// GAT_10599979287264
// MI455X (gfx1250) — matrix operand order repaired and checked by reading, not compiled, not run on hardware
//
#include <hip/hip_runtime.h>

#define NODES 6144
#define FIN   128
#define HEADS 4
#define FOUT  64
#define OUTC  (HEADS*FOUT)   // 256

typedef _Float16 v16h __attribute__((ext_vector_type(16)));
typedef _Float16 v8h  __attribute__((ext_vector_type(8)));
typedef float    v8f  __attribute__((ext_vector_type(8)));
typedef int      vi4  __attribute__((ext_vector_type(4)));

#if defined(__has_builtin)
#if __has_builtin(__builtin_amdgcn_global_load_async_to_lds_b128)
#define ASYNC_LDS 1
#endif
#endif

#if defined(ASYNC_LDS)
#if __has_builtin(__builtin_amdgcn_s_wait_asynccnt)
#define WAIT_ASYNC(n) __builtin_amdgcn_s_wait_asynccnt(n)
#else
#define WAIT_ASYNC(n) asm volatile("s_wait_asynccnt %0" :: "n"(n) : "memory")
#endif
#endif

static __device__ __forceinline__ v16h cat8(v8h lo, v8h hi) {
    return __builtin_shufflevector(lo, hi, 0,1,2,3,4,5,6,7,8,9,10,11,12,13,14,15);
}
static __device__ __forceinline__ float lrelu(float x) { return fmaxf(x, 0.2f * x); }
static __device__ __forceinline__ float fast_exp(float x) {
    return __builtin_amdgcn_exp2f(x * 1.4426950408889634f);
}

// ---------------------------------------------------------------------------
// Kernel 0: convert x -> f16, build W^T (f16) as wT[(h*64+o)*128 + f]
// ---------------------------------------------------------------------------
__global__ void k_prep(const float* __restrict__ x, const float* __restrict__ W,
                       _Float16* __restrict__ xh, _Float16* __restrict__ wT) {
    int stride = gridDim.x * blockDim.x;
    for (int i = blockIdx.x * blockDim.x + threadIdx.x; i < NODES * FIN; i += stride)
        xh[i] = (_Float16)x[i];
    for (int i = blockIdx.x * blockDim.x + threadIdx.x; i < HEADS * FIN * FOUT; i += stride) {
        int hh  = i >> 13;        // / (128*64)
        int rem = i & 8191;
        int o   = rem >> 7;       // / 128
        int f   = rem & 127;
        wT[i] = (_Float16)W[hh * FIN * FOUT + f * FOUT + o];
    }
}

// ---------------------------------------------------------------------------
// Kernel 1: proj = x @ W per head via WMMA; emit projT (f16, [h][o][node]),
// s_src[h][n], s_tgt[h][n] (fp32). Block = 8 waves: 2 i-tiles x 4 heads.
// ---------------------------------------------------------------------------
__global__ __launch_bounds__(256) void k_proj(const _Float16* __restrict__ xh,
        const _Float16* __restrict__ wT,
        const float* __restrict__ a_src, const float* __restrict__ a_tgt,
        _Float16* __restrict__ projT,
        float* __restrict__ s_src, float* __restrict__ s_tgt) {
    const int lane = threadIdx.x & 31;
    const int w    = threadIdx.x >> 5;
    const int h    = w & 3;
    const int it   = blockIdx.x * 2 + (w >> 2);
    const int m16  = lane & 15;
    const int hi   = lane >> 4;
    const int kb   = hi * 8;

    v8f acc[4] = {};
    const int row = it * 16 + m16;

    #pragma unroll
    for (int ks = 0; ks < 4; ++ks) {
        const int k0 = ks * 32;
        const _Float16* pa = xh + row * FIN + k0 + kb;
        v16h A = cat8(*(const v8h*)pa, *(const v8h*)(pa + 16));
        #pragma unroll
        for (int nt = 0; nt < 4; ++nt) {
            const int n = nt * 16 + m16;
            const _Float16* pb = wT + (h * FOUT + n) * FIN + k0 + hi * 8;
            v16h B = cat8(*(const v8h*)pb, *(const v8h*)(pb + 16));
            acc[nt] = __builtin_amdgcn_wmma_f32_16x16x32_f16(
                false, A, false, B, (short)0, acc[nt], false, false);
        }
    }

    #pragma unroll
    for (int nt = 0; nt < 4; ++nt) {
        const int n = nt * 16 + m16;
        #pragma unroll
        for (int r = 0; r < 8; ++r) {
            projT[(h * FOUT + n) * NODES + it * 16 + r + 8 * hi] = (_Float16)acc[nt][r];
        }
    }

    float aS[4], aT[4];
    #pragma unroll
    for (int nt = 0; nt < 4; ++nt) {
        aS[nt] = a_src[h * FOUT + nt * 16 + m16];
        aT[nt] = a_tgt[h * FOUT + nt * 16 + m16];
    }
    #pragma unroll
    for (int r = 0; r < 8; ++r) {
        float ps = 0.f, pt = 0.f;
        #pragma unroll
        for (int nt = 0; nt < 4; ++nt) {
            ps += acc[nt][r] * aS[nt];
            pt += acc[nt][r] * aT[nt];
        }
        #pragma unroll
        for (int d = 1; d < 16; d <<= 1) {
            ps += __shfl_xor(ps, d, 32);
            pt += __shfl_xor(pt, d, 32);
        }
        if (m16 == 0) {
            const int orow = it * 16 + r + 8 * hi;
            s_src[h * NODES + orow] = ps;
            s_tgt[h * NODES + orow] = pt;
        }
    }
}

// ---------------------------------------------------------------------------
// Kernel T: T[h] = max_j s_tgt[h][j]
// ---------------------------------------------------------------------------
__global__ __launch_bounds__(256) void k_tmax(const float* __restrict__ s_tgt,
                                              float* __restrict__ Tm) {
    __shared__ float red[256];
    const int h = blockIdx.x;
    float m = -1e30f;
    for (int i = threadIdx.x; i < NODES; i += 256) m = fmaxf(m, s_tgt[h * NODES + i]);
    red[threadIdx.x] = m;
    __syncthreads();
    for (int s = 128; s > 0; s >>= 1) {
        if (threadIdx.x < s) red[threadIdx.x] = fmaxf(red[threadIdx.x], red[threadIdx.x + s]);
        __syncthreads();
    }
    if (threadIdx.x == 0) Tm[h] = red[0];
}

// ---------------------------------------------------------------------------
// Kernel 2: fused masked-softmax attention + attn@proj (flash-style, single
// pass, analytic shift m_i = lrelu(s_src[i] + max_j s_tgt[j])).
// Block = 128 threads = 4 waves; wave w = head w, shared 16-row i-tile.
// Double-buffered LDS staging; async global->LDS copies when available.
// ---------------------------------------------------------------------------
__global__ __launch_bounds__(128) void k_attn(const float* __restrict__ mask,
        const _Float16* __restrict__ projT,
        const float* __restrict__ s_src, const float* __restrict__ s_tgt,
        const float* __restrict__ Tm, const float* __restrict__ bias,
        float* __restrict__ out) {
    __shared__ __align__(16) _Float16 ldsV[2][HEADS * FOUT * 32]; // V tiles [h][o][k]
    __shared__ __align__(16) float    ldsM[2][16 * 32];           // mask tiles [row][k]

    const int t    = threadIdx.x;
    const int lane = t & 31;
    const int h    = t >> 5;       // head = wave id
    const int m16  = lane & 15;
    const int hi   = lane >> 4;
    const int kb   = hi * 8;
    const int i0   = blockIdx.x * 16;

    const float ssrc = s_src[h * NODES + i0 + m16];
    const float mrow = lrelu(ssrc + Tm[h]);
    float rsum = 0.f;
    v8f acc[4] = {};

    const int mr  = t >> 3;            // staging: mask row 0..15
    const int mc4 = (t & 7) * 4;       // staging: mask col group

#if defined(ASYNC_LDS)
    // issue 9 async b128 copies per thread into buffer b for tile at column jn
    auto stage = [&](int b, int jn) {
        __builtin_amdgcn_global_load_async_to_lds_b128(
            (vi4*)&mask[(size_t)(i0 + mr) * NODES + jn + mc4],
            (vi4*)&ldsM[b][mr * 32 + mc4], 0, 0);
        #pragma unroll
        for (int q2 = 0; q2 < 2; ++q2) {
            const int q = t * 2 + q2;                  // q = h*64+o
            const _Float16* src = projT + q * NODES + jn;
            _Float16* dst = &ldsV[b][q * 32];
            #pragma unroll
            for (int c = 0; c < 4; ++c)
                __builtin_amdgcn_global_load_async_to_lds_b128(
                    (vi4*)(src + c * 8), (vi4*)(dst + c * 8), 0, 0);
        }
    };
#else
    auto stage = [&](int b, int jn) {
        *(float4*)&ldsM[b][mr * 32 + mc4] =
            *(const float4*)&mask[(size_t)(i0 + mr) * NODES + jn + mc4];
        #pragma unroll
        for (int q2 = 0; q2 < 2; ++q2) {
            const int q = t * 2 + q2;
            const float4* src = (const float4*)(projT + q * NODES + jn);
            float4* dst = (float4*)&ldsV[b][q * 32];
            dst[0] = src[0]; dst[1] = src[1]; dst[2] = src[2]; dst[3] = src[3];
        }
    };
#endif

    const int NT = NODES / 32;
    int cur = 0;
#if defined(ASYNC_LDS)
    stage(0, 0);                                   // prologue: fill buffer 0
#endif

    for (int jt = 0; jt < NT; ++jt) {
        const int j = jt * 32;
#if defined(ASYNC_LDS)
        if (jt + 1 < NT) { stage(cur ^ 1, j + 32); WAIT_ASYNC(9); }
        else             { WAIT_ASYNC(0); }
        __syncthreads();                           // all waves' tile data landed
#else
        __syncthreads();
        stage(cur, j);
        __syncthreads();
#endif

        const float* stp = &s_tgt[h * NODES + j];
        float4 st0 = *(const float4*)(stp + kb);
        float4 st1 = *(const float4*)(stp + kb + 4);
        float4 st2 = *(const float4*)(stp + kb + 16);
        float4 st3 = *(const float4*)(stp + kb + 20);
        float4 mk0 = *(const float4*)&ldsM[cur][m16 * 32 + kb];
        float4 mk1 = *(const float4*)&ldsM[cur][m16 * 32 + kb + 4];
        float4 mk2 = *(const float4*)&ldsM[cur][m16 * 32 + kb + 16];
        float4 mk3 = *(const float4*)&ldsM[cur][m16 * 32 + kb + 20];
        const float stv[16] = {st0.x, st0.y, st0.z, st0.w, st1.x, st1.y, st1.z, st1.w,
                               st2.x, st2.y, st2.z, st2.w, st3.x, st3.y, st3.z, st3.w};
        const float mkv[16] = {mk0.x, mk0.y, mk0.z, mk0.w, mk1.x, mk1.y, mk1.z, mk1.w,
                               mk2.x, mk2.y, mk2.z, mk2.w, mk3.x, mk3.y, mk3.z, mk3.w};

        v16h A;
        #pragma unroll
        for (int e = 0; e < 16; ++e) {
            const float sc = lrelu(ssrc + stv[e]) + mkv[e];
            const float p  = fast_exp(sc - mrow);
            rsum += p;
            A[e] = (_Float16)p;
        }

        #pragma unroll
        for (int nt = 0; nt < 4; ++nt) {
            const _Float16* pb = &ldsV[cur][(h * FOUT + nt * 16 + m16) * 32 + hi * 8];
            v16h B = cat8(*(const v8h*)pb, *(const v8h*)(pb + 16));
            acc[nt] = __builtin_amdgcn_wmma_f32_16x16x32_f16(
                false, A, false, B, (short)0, acc[nt], false, false);
        }

#if defined(ASYNC_LDS)
        __syncthreads();            // all waves done reading 'cur' before reuse
        cur ^= 1;
#endif
    }

    rsum += __shfl_xor(rsum, 16, 32);
    const float inv = __builtin_amdgcn_rcpf(rsum);

    #pragma unroll
    for (int r = 0; r < 8; ++r) {
        const float invr = __shfl(inv, r + 8 * hi, 32);
        const int orow = i0 + r + 8 * hi;
        #pragma unroll
        for (int nt = 0; nt < 4; ++nt) {
            const int oc = h * FOUT + nt * 16 + m16;
            float v = acc[nt][r] * invr + bias[oc];
            v = (v > 0.f) ? v : (fast_exp(v) - 1.f);
            out[(size_t)orow * OUTC + oc] = v;
        }
    }
}

// ---------------------------------------------------------------------------
extern "C" void kernel_launch(void* const* d_in, const int* in_sizes, int n_in,
                              void* d_out, int out_size, void* d_ws, size_t ws_size,
                              hipStream_t stream) {
    const float* x    = (const float*)d_in[0];   // [6144,128]
    const float* mask = (const float*)d_in[1];   // [6144,6144]
    const float* W    = (const float*)d_in[2];   // [4,128,64]
    const float* a_s  = (const float*)d_in[3];   // [4,64,1]
    const float* a_t  = (const float*)d_in[4];   // [4,64,1]
    const float* bias = (const float*)d_in[5];   // [256]
    float* out = (float*)d_out;                  // [6144,256]

    char* ws = (char*)d_ws;
    _Float16* xh    = (_Float16*)(ws);               // 1,572,864 B
    _Float16* wT    = (_Float16*)(ws + 1572864);     //    65,536 B
    _Float16* projT = (_Float16*)(ws + 1638400);     // 3,145,728 B
    float*    s_src = (float*)   (ws + 4784128);     //    98,304 B
    float*    s_tgt = (float*)   (ws + 4882432);     //    98,304 B
    float*    Tm    = (float*)   (ws + 4980736);     //        16 B

    k_prep<<<1024, 256, 0, stream>>>(x, W, xh, wT);
    k_proj<<<NODES / 32, 256, 0, stream>>>(xh, wT, a_s, a_t, projT, s_src, s_tgt);
    k_tmax<<<HEADS, 256, 0, stream>>>(s_tgt, Tm);
    k_attn<<<NODES / 16, 128, 0, stream>>>(mask, projT, s_src, s_tgt, Tm, bias, out);
}
